// MultiScale_gcn_18743237280146
// MI455X (gfx1250) — matrix operand order repaired and checked by reading, not compiled, not run on hardware
//
#include <hip/hip_runtime.h>
#include <hip/hip_bf16.h>

// ---------------------------------------------------------------------------
// MultiScale GCN fused pipeline for gfx1250 (MI455X), bf16 WMMA + f32 accum.
//   k0: zero padded A_full workspace (pads must be clean zeros for WMMA K-pad)
//   k1: A_full[s,n,o,v,w] = (ada_w[s] @ ada_A[n] + ada_b) * PA -> ws bf16 [32x32]
//   k2: per (n, o-block16, t-tile16): feat GEMM -> LDS, per-channel graph matmul
//       (A_full tiles staged via GLOBAL_LOAD_ASYNC_TO_LDS_B128, latency hidden
//       behind stage-1 WMMAs), residual GEMM, fused BN+BN+ReLU epilogue.
// ---------------------------------------------------------------------------

typedef __attribute__((ext_vector_type(16))) __bf16 v16bf;
typedef __attribute__((ext_vector_type(8)))  __bf16 v8bf;
typedef __attribute__((ext_vector_type(8)))  float  v8f;

#define N_    32
#define CIN_  64
#define COUT_ 128
#define V_    25
#define T_    300
#define S_    3
#define EPS_  1e-5f

// A_full workspace: [S][N][COUT][32][32] bf16 (v,w padded 25->32)
#define WS_A_ELEMS ((size_t)S_ * N_ * COUT_ * 32 * 32)

union V16U { v16bf v; v8bf h[2]; };

// A-operand (16x32 bf16, MxK) per-lane element->K mapping (ISA 7.12.2)
__device__ __forceinline__ int akmap(int lane, int j) {
  return ((j >> 3) << 4) + ((lane >> 4) << 3) + (j & 7);
}

// ---------------------------------------------------------------------------
__global__ void k0_zero_ws(unsigned int* __restrict__ ws, int nwords) {
  int i = blockIdx.x * blockDim.x + threadIdx.x;
  int stride = gridDim.x * blockDim.x;
  for (; i < nwords; i += stride) ws[i] = 0u;
}

// ---------------------------------------------------------------------------
// k1: per (s,n): adaA[o, col=(v*25+w)] = ada_w[s] @ ada_A[n], then (+ bias) * PA.
__global__ __launch_bounds__(256) void k1_build_A(
    const float* __restrict__ ada_A,  // [N,CIN,V,V]
    const float* __restrict__ PA,     // [S,V,V]
    const float* __restrict__ ada_w,  // [S,COUT,CIN]
    const float* __restrict__ ada_b,  // [S,COUT]
    __bf16* __restrict__ wsA)         // [S,N,COUT,32,32]
{
  __attribute__((aligned(32))) __shared__ __bf16 ads[640][64];  // [col][i], cols padded 625->640

  const int s    = blockIdx.x / N_;
  const int n    = blockIdx.x % N_;
  const int tid  = threadIdx.x;
  const int lane = tid & 31;
  const int wv   = tid >> 5;        // 8 waves -> 8 o-tiles of 16

  // stage ada_A[n] into LDS as bf16, B-operand friendly [col][i]
  const float* aA = ada_A + (size_t)n * CIN_ * V_ * V_;
  for (int idx = tid; idx < 640 * CIN_; idx += 256) {
    int i   = idx / 640;
    int col = idx % 640;
    float vf = (col < V_ * V_) ? aA[(size_t)i * (V_ * V_) + col] : 0.f;
    ads[col][i] = (__bf16)vf;
  }
  __syncthreads();

  // A operand: 16 rows o = 16*wv + (lane&15), K = i (2 K-tiles of 32)
  const int obase = wv * 16;
  const float* W = ada_w + ((size_t)s * COUT_ + obase) * CIN_;
  V16U aw[2];
  #pragma unroll
  for (int kt = 0; kt < 2; ++kt)
    #pragma unroll
    for (int j = 0; j < 16; ++j)
      aw[kt].v[j] = (__bf16)W[(size_t)(lane & 15) * CIN_ + kt * 32 + akmap(lane, j)];

  // hoist per-lane bias values (loop-invariant)
  const float* bias = ada_b + s * COUT_;
  float biasv[8];
  #pragma unroll
  for (int r = 0; r < 8; ++r) biasv[r] = bias[obase + r + ((lane >> 4) << 3)];

  __bf16* wsblk = wsA + ((size_t)s * N_ + n) * COUT_ * 1024;

  for (int ct = 0; ct < 40; ++ct) {          // 640 cols / 16
    v8f acc = {};
    #pragma unroll
    for (int kt = 0; kt < 2; ++kt) {
      V16U b;
      const __bf16* bp = &ads[ct * 16 + (lane & 15)][kt * 32 + ((lane >> 4) << 3)];
      b.h[0] = *(const v8bf*)bp;
      b.h[1] = *(const v8bf*)(bp + 16);
      acc = __builtin_amdgcn_wmma_f32_16x16x32_bf16(false, aw[kt].v, false, b.v,
                                                    (short)0, acc, false, false);
    }
    int col = ct * 16 + (lane & 15);
    if (col < V_ * V_) {
      int v = col / V_, w = col % V_;
      float pa = PA[((size_t)s * V_ + v) * V_ + w];
      #pragma unroll
      for (int r = 0; r < 8; ++r) {
        int o = obase + r + ((lane >> 4) << 3);
        float val = (acc[r] + biasv[r]) * pa;
        wsblk[(size_t)o * 1024 + v * 32 + w] = (__bf16)val;
      }
    }
  }
}

// ---------------------------------------------------------------------------
// k2: fused feat GEMM + graph matmul + residual + BN/BN/ReLU.
// grid: (t-tiles=19, o-blocks=8, n=32), 256 threads (8 waves).
__global__ __launch_bounds__(256) void k2_fused(
    const float* __restrict__ x,        // [N,CIN,V,T]
    const float* __restrict__ conv3_w,  // [S,COUT,CIN]
    const float* __restrict__ conv3_b,  // [S,COUT]
    const float* __restrict__ down_w,   // [COUT,CIN]
    const float* __restrict__ down_b,   // [COUT]
    const float* __restrict__ bn_g,  const float* __restrict__ bn_b,
    const float* __restrict__ bn_m,  const float* __restrict__ bn_v,
    const float* __restrict__ dbn_g, const float* __restrict__ dbn_b,
    const float* __restrict__ dbn_m, const float* __restrict__ dbn_v,
    const __bf16* __restrict__ wsA,     // [S,N,COUT,32,32]
    float* __restrict__ out)            // [N,COUT,V,T]
{
  // x tile, bf16, [w][t][i] (i padded 64->72: 144B rows, 16B aligned)
  __attribute__((aligned(32))) __shared__ __bf16 xs[V_][16][72];     // 57.6 KB
  // feat / residual buffer, bf16, [o_local][t][w] (w padded 32->40: 80B rows)
  __attribute__((aligned(32))) __shared__ __bf16 fb[16][16][40];     // 20.5 KB
  // async-staged A_full rows for this block's 16 output channels
  __attribute__((aligned(32))) __shared__ __bf16 aslds[16][1024];    // 32 KB

  const int tBase = blockIdx.x * 16;
  const int OB    = blockIdx.y * 16;
  const int n     = blockIdx.z;
  const int tid   = threadIdx.x;
  const int lane  = tid & 31;
  const int wv    = tid >> 5;

  // cooperative x tile load (t-tail zero padded)
  for (int idx = tid; idx < V_ * 16 * CIN_; idx += 256) {
    int t = idx & 15;
    int w = (idx >> 4) % V_;
    int i = idx / (16 * V_);
    int tg = tBase + t;
    float vf = (tg < T_) ? x[(((size_t)n * CIN_ + i) * V_ + w) * T_ + tg] : 0.f;
    xs[w][t][i] = (__bf16)vf;
  }
  // zero fb K-padding rows w in [25,32) (stage-2 reads K up to 32)
  for (int idx = tid; idx < 16 * 16 * 7; idx += 256) {
    int w = 25 + (idx % 7);
    int t = (idx / 7) & 15;
    int o = idx / (7 * 16);
    fb[o][t][w] = (__bf16)0.f;
  }
  __syncthreads();

  v8f acc2[2][2] = {};   // [o per wave][v-tile]

  // passes 0..2 = subsets (feat + graph matmul), pass 3 = residual conv
  for (int p = 0; p < 4; ++p) {
    // ---- async-stage this wave's two A_full rows (4 KB, wave-private) ----
    // Latency is hidden behind stage-1 WMMA work; waited before stage 2.
    if (p < 3) {
      const __bf16* wsrow = wsA + (((size_t)p * N_ + n) * COUT_ + OB + wv * 2) * 1024;
      unsigned int ldsb = (unsigned int)(unsigned long long)(&aslds[wv * 2][0])
                          + (unsigned int)(lane * 16);
      const char* gsrc = (const char*)wsrow + lane * 16;
      #pragma unroll
      for (int k = 0; k < 8; ++k) {
        // INST_OFFSET is added to both the LDS and global address (ISA 10.x):
        // one VGPR address pair serves all 8 x 512B chunks.
        asm volatile("global_load_async_to_lds_b128 %0, %1, off offset:%2"
                     :: "v"(ldsb), "v"(gsrc), "i"(k * 512) : "memory");
      }
    }

    const float* W = (p < 3) ? conv3_w + (size_t)p * COUT_ * CIN_ : down_w;
    const float* B = (p < 3) ? conv3_b + p * COUT_ : down_b;

    // weight A-operand tiles for rows OB..OB+15
    V16U aw[2];
    #pragma unroll
    for (int kt = 0; kt < 2; ++kt)
      #pragma unroll
      for (int j = 0; j < 16; ++j)
        aw[kt].v[j] = (__bf16)W[(size_t)(OB + (lane & 15)) * CIN_ + kt * 32 + akmap(lane, j)];
    float biasv[8];
    #pragma unroll
    for (int r = 0; r < 8; ++r) biasv[r] = B[OB + r + ((lane >> 4) << 3)];

    if (p > 0) __syncthreads();   // previous stage-2 done reading fb

    // ---- stage 1: [16 o] x [K=64 i] x [25w x 16t cols] GEMM into fb ----
    for (int w = wv; w < V_; w += 8) {
      int t = lane & 15;
      v8f acc = {};
      #pragma unroll
      for (int kt = 0; kt < 2; ++kt) {
        V16U b;
        const __bf16* bp = &xs[w][t][kt * 32 + ((lane >> 4) << 3)];
        b.h[0] = *(const v8bf*)bp;
        b.h[1] = *(const v8bf*)(bp + 16);
        acc = __builtin_amdgcn_wmma_f32_16x16x32_bf16(false, aw[kt].v, false, b.v,
                                                      (short)0, acc, false, false);
      }
      #pragma unroll
      for (int r = 0; r < 8; ++r) {
        int m = r + ((lane >> 4) << 3);
        fb[m][t][w] = (__bf16)(acc[r] + biasv[r]);
      }
    }
    __syncthreads();

    // ---- stage 2: D[v,t] += A_full[p,n,o] @ feat (per-channel 32x16, K=32) ----
    if (p < 3) {
      asm volatile("s_wait_asynccnt 0x0" ::: "memory");  // A rows landed in LDS
      #pragma unroll
      for (int oi = 0; oi < 2; ++oi) {
        int oL = wv * 2 + oi;
        #pragma unroll
        for (int vt = 0; vt < 2; ++vt) {
          V16U a;  // A operand: M=v, K=w (from async-staged LDS)
          const __bf16* ap = &aslds[oL][(vt * 16 + (lane & 15)) * 32 + ((lane >> 4) << 3)];
          a.h[0] = *(const v8bf*)ap;
          a.h[1] = *(const v8bf*)(ap + 16);
          V16U b;  // B operand: K=w, N=t
          const __bf16* bp = &fb[oL][lane & 15][(lane >> 4) << 3];
          b.h[0] = *(const v8bf*)bp;
          b.h[1] = *(const v8bf*)(bp + 16);
          acc2[oi][vt] = __builtin_amdgcn_wmma_f32_16x16x32_bf16(
              false, a.v, false, b.v, (short)0, acc2[oi][vt], false, false);
        }
      }
    }
  }

  // ---- epilogue: BN(main) + BN(residual from fb) + ReLU ----
  #pragma unroll
  for (int oi = 0; oi < 2; ++oi) {
    int oL = wv * 2 + oi;
    int o  = OB + oL;
    float bs  = bn_g[o]  * rsqrtf(bn_v[o]  + EPS_);
    float bsh = bn_b[o]  - bn_m[o]  * bs;
    float ds  = dbn_g[o] * rsqrtf(dbn_v[o] + EPS_);
    float dsh = dbn_b[o] - dbn_m[o] * ds;
    int t  = lane & 15;
    int tg = tBase + t;
    #pragma unroll
    for (int vt = 0; vt < 2; ++vt) {
      #pragma unroll
      for (int r = 0; r < 8; ++r) {
        int v = vt * 16 + r + ((lane >> 4) << 3);
        if (v < V_ && tg < T_) {
          float y  = acc2[oi][vt][r] * bs + bsh;
          float rr = (float)fb[oL][t][v] * ds + dsh;
          float z  = y + rr;
          out[(((size_t)n * COUT_ + o) * V_ + v) * T_ + tg] = z > 0.f ? z : 0.f;
        }
      }
    }
  }
}

// ---------------------------------------------------------------------------
extern "C" void kernel_launch(void* const* d_in, const int* in_sizes, int n_in,
                              void* d_out, int out_size, void* d_ws, size_t ws_size,
                              hipStream_t stream) {
  (void)in_sizes; (void)n_in; (void)out_size; (void)ws_size;

  const float* x       = (const float*)d_in[0];
  const float* ada_A   = (const float*)d_in[1];
  const float* PA      = (const float*)d_in[2];
  const float* conv3_w = (const float*)d_in[3];
  const float* conv3_b = (const float*)d_in[4];
  const float* ada_w   = (const float*)d_in[5];
  const float* ada_b   = (const float*)d_in[6];
  const float* bn_g    = (const float*)d_in[7];
  const float* bn_b    = (const float*)d_in[8];
  const float* bn_m    = (const float*)d_in[9];
  const float* bn_v    = (const float*)d_in[10];
  const float* down_w  = (const float*)d_in[11];
  const float* down_b  = (const float*)d_in[12];
  const float* dbn_g   = (const float*)d_in[13];
  const float* dbn_b   = (const float*)d_in[14];
  const float* dbn_m   = (const float*)d_in[15];
  const float* dbn_v   = (const float*)d_in[16];

  __bf16* wsA = (__bf16*)d_ws;   // needs WS_A_ELEMS*2 = ~25.2 MB

  // k0: zero padded A_full workspace (2 bf16 per u32)
  int nwords = (int)(WS_A_ELEMS / 2);
  k0_zero_ws<<<1024, 256, 0, stream>>>((unsigned int*)d_ws, nwords);

  // k1: build A_full (bf16, padded)
  k1_build_A<<<S_ * N_, 256, 0, stream>>>(ada_A, PA, ada_w, ada_b, wsA);

  // k2: fused main pass
  dim3 grid((T_ + 15) / 16, COUT_ / 16, N_);
  k2_fused<<<grid, 256, 0, stream>>>(x, conv3_w, conv3_b, down_w, down_b,
                                     bn_g, bn_b, bn_m, bn_v,
                                     dbn_g, dbn_b, dbn_m, dbn_v,
                                     wsA, (float*)d_out);
}
